// WanJointTokenAttention_85564338471357
// MI455X (gfx1250) — compile-verified, matrix operand order checked by reading, not run on hardware
//
#include <hip/hip_runtime.h>

// ---------------- problem constants ----------------
#define Bdim 4
#define Cdim 3072
#define Tdim 4096
#define KJ 24          // joints (M dimension, padded to 32)
#define CPJd 128       // channels per joint
#define TDd 128        // token dim
#define TT 4           // t-positions per workgroup
#define NTHREADS 256   // 8 wave32
#define LDA 132        // row stride (u16) for 128-col LDS tiles (pad vs bank conflicts)
#define LDV 36         // row stride (u16) for v-transposed [128][24->32 pad]
#define LDATT 36       // row stride (u16) for attention weights [32][32 pad]
#define LDSC 28        // row stride (f32) for raw scores
#define EPSf 1e-5f
#define SCALEf 0.08838834764831845f   // 128^-0.5

typedef __attribute__((ext_vector_type(16))) __bf16 v16bf;
typedef __attribute__((ext_vector_type(8)))  float  v8f;

union FragU { v16bf v; unsigned int u[8]; };

__device__ __forceinline__ unsigned short f2bf(float f) {
  unsigned int u = __float_as_uint(f);
  u += 0x7FFFu + ((u >> 16) & 1u);   // round-to-nearest-even
  return (unsigned short)(u >> 16);
}

// 16x32 bf16 fragment (A layout; B uses same striping with lane<->N when the
// matrix is stored transposed [n][k]).  Per ISA 7.12.2: lanes 0-15 hold rows,
// lanes 16-31 the K+8 half; VGPR v holds K pair (v&3)*2 + (v>>2)*16.
__device__ __forceinline__ v16bf load_frag(const unsigned short* buf, int ld,
                                           int row0, int col0, int lane) {
  FragU fu;
  const int m  = row0 + (lane & 15);
  const int kh = (lane >> 4) * 8;
  const unsigned short* p = buf + m * ld + col0 + kh;
#pragma unroll
  for (int v = 0; v < 8; ++v) {
    const int k0 = (v & 3) * 2 + (v >> 2) * 16;
    fu.u[v] = *(const unsigned int*)(p + k0);   // dword-aligned (all offsets even)
  }
  return fu.v;
}

__device__ __forceinline__ v8f wmma_bf16(v16bf a, v16bf b, v8f c) {
  return __builtin_amdgcn_wmma_f32_16x16x32_bf16(false, a, false, b, (short)0, c,
                                                 false, false);
}

// Stage one 128x128 fp32 weight chunk -> LDS, transposed, bf16.
// Global reads are coalesced along n (row-major W[kd][col0+n]).
__device__ __forceinline__ void stage_w(const float* __restrict__ W, int ldw,
                                        int col0, unsigned short* dst, int tid) {
#pragma unroll 4
  for (int i = tid; i < 128 * 128; i += NTHREADS) {
    const int n  = i & 127;
    const int kd = i >> 7;
    dst[n * LDA + kd] = f2bf(W[(size_t)kd * ldw + col0 + n]);
  }
}

// 32-tile GEMM: C[2 Mtiles][4 Ntiles] += A[24x128] * Bt[128x128]^T-layout
__device__ __forceinline__ void mm24(const unsigned short* Ab, int lda,
                                     const unsigned short* Bt, int ldb,
                                     int nbase, int lane, v8f acc[2][4]) {
#pragma unroll
  for (int kk = 0; kk < 4; ++kk) {
    v16bf a0 = load_frag(Ab, lda, 0,  kk * 32, lane);
    v16bf a1 = load_frag(Ab, lda, 16, kk * 32, lane);
#pragma unroll
    for (int nt = 0; nt < 4; ++nt) {
      v16bf bb = load_frag(Bt, ldb, (nbase + nt) * 16, kk * 32, lane);
      acc[0][nt] = wmma_bf16(a0, bb, acc[0][nt]);
      acc[1][nt] = wmma_bf16(a1, bb, acc[1][nt]);
    }
  }
}

__global__ __launch_bounds__(NTHREADS, 1)
void wan_joint_attn_kernel(const float* __restrict__ x,
                           const float* __restrict__ W_tok,
                           const float* __restrict__ b_tok,
                           const float* __restrict__ g_norm,
                           const float* __restrict__ W_qkv,
                           const float* __restrict__ b_qkv,
                           const float* __restrict__ W_proj,
                           const float* __restrict__ b_proj,
                           const float* __restrict__ W_from,
                           const float* __restrict__ b_from,
                           float* __restrict__ out) {
  __shared__ unsigned short wbuf[128 * LDA];          // shared weight stage (bf16, transposed)
  __shared__ unsigned short bufA[TT][32 * LDA];       // x slab -> tok(normed) -> attn@v out
  __shared__ unsigned short qb[TT][32 * LDA];         // q -> proj result
  __shared__ unsigned short kb[TT][32 * LDA];         // k
  __shared__ unsigned short vT[TT][TDd * LDV];        // v transposed [d][j]
  __shared__ unsigned short attnb[TT][32 * LDATT];    // softmaxed attention (bf16)
  __shared__ float          sbuf[TT][KJ * LDSC];      // raw scores (f32)
  __shared__ float          rmsbuf[TT][2][32];        // cross-wave sum-of-squares

  const int tid  = threadIdx.x;
  const int lane = tid & 31;
  const int wave = tid >> 5;
  const int pos  = wave >> 1;      // which of the 4 t-positions
  const int nh   = wave & 1;       // which N-half (tiles 0..3 vs 4..7)
  const int lrow = lane & 15;
  const int lhi  = lane >> 4;

  const int bB = blockIdx.x / (Tdim / TT);
  const int t0 = (blockIdx.x % (Tdim / TT)) * TT;
  const int t  = t0 + pos;

  // ---------------- Phase 0: stage W_tok, zero vT pad, load x slab ----------
  stage_w(W_tok, TDd, 0, wbuf, tid);
  {
    unsigned int* vz = (unsigned int*)&vT[0][0];
    for (int i = tid; i < TT * TDd * LDV / 2; i += NTHREADS) vz[i] = 0u;
  }
  for (int i = tid; i < KJ * CPJd; i += NTHREADS) {
    const int c = i & 127;
    const int j = i >> 7;
    const float4 xv =
        *(const float4*)(x + ((size_t)(bB * Cdim + j * CPJd + c)) * Tdim + t0);
    const int o = j * LDA + c;
    bufA[0][o] = f2bf(xv.x);
    bufA[1][o] = f2bf(xv.y);
    bufA[2][o] = f2bf(xv.z);
    bufA[3][o] = f2bf(xv.w);
  }
  __syncthreads();

  const v8f vzero = {0.f, 0.f, 0.f, 0.f, 0.f, 0.f, 0.f, 0.f};
  v8f acc[2][4];

  // ---------------- tok = x_slab @ W_tok + b_tok, then RMSNorm --------------
#pragma unroll
  for (int mt = 0; mt < 2; ++mt)
#pragma unroll
    for (int nt = 0; nt < 4; ++nt) acc[mt][nt] = vzero;
  mm24(bufA[pos], LDA, wbuf, LDA, nh * 4, lane, acc);

#pragma unroll
  for (int mt = 0; mt < 2; ++mt) {
#pragma unroll
    for (int nt = 0; nt < 4; ++nt) {
      const int gcol = (nh * 4 + nt) * 16 + lrow;
      const float bias = b_tok[gcol];
#pragma unroll
      for (int r = 0; r < 8; ++r) acc[mt][nt][r] += bias;
    }
#pragma unroll
    for (int r = 0; r < 8; ++r) {
      float p = 0.f;
#pragma unroll
      for (int nt = 0; nt < 4; ++nt) { const float v = acc[mt][nt][r]; p += v * v; }
      p += __shfl_xor(p, 1, 16);
      p += __shfl_xor(p, 2, 16);
      p += __shfl_xor(p, 4, 16);
      p += __shfl_xor(p, 8, 16);
      if (lrow == 0) rmsbuf[pos][nh][mt * 16 + r + 8 * lhi] = p;
    }
  }
  __syncthreads();

#pragma unroll
  for (int mt = 0; mt < 2; ++mt) {
    float sr[8];
#pragma unroll
    for (int r = 0; r < 8; ++r) {
      const int row = mt * 16 + r + 8 * lhi;
      const float tot = rmsbuf[pos][0][row] + rmsbuf[pos][1][row];
      sr[r] = rsqrtf(tot * (1.0f / TDd) + EPSf);
    }
#pragma unroll
    for (int nt = 0; nt < 4; ++nt) {
      const int gcol = (nh * 4 + nt) * 16 + lrow;
      const float gn = g_norm[gcol];
#pragma unroll
      for (int r = 0; r < 8; ++r) {
        const int row = mt * 16 + r + 8 * lhi;
        if (row < KJ) bufA[pos][row * LDA + gcol] = f2bf(acc[mt][nt][r] * sr[r] * gn);
      }
    }
  }

  // ---------------- qkv: three 128-col chunks through the shared wbuf -------
#pragma unroll 1
  for (int ch = 0; ch < 3; ++ch) {
    __syncthreads();                       // wbuf free, tok writes visible
    stage_w(W_qkv, 3 * TDd, ch * TDd, wbuf, tid);
    __syncthreads();
#pragma unroll
    for (int mt = 0; mt < 2; ++mt)
#pragma unroll
      for (int nt = 0; nt < 4; ++nt) acc[mt][nt] = vzero;
    mm24(bufA[pos], LDA, wbuf, LDA, nh * 4, lane, acc);
#pragma unroll
    for (int mt = 0; mt < 2; ++mt)
#pragma unroll
      for (int nt = 0; nt < 4; ++nt) {
        const int gcol = (nh * 4 + nt) * 16 + lrow;
        const float bias = b_qkv[ch * TDd + gcol];
#pragma unroll
        for (int r = 0; r < 8; ++r) {
          const int row = mt * 16 + r + 8 * lhi;
          if (row < KJ) {
            const unsigned short us = f2bf(acc[mt][nt][r] + bias);
            if (ch == 0)      qb[pos][row * LDA + gcol] = us;
            else if (ch == 1) kb[pos][row * LDA + gcol] = us;
            else              vT[pos][gcol * LDV + row] = us;   // transposed
          }
        }
      }
  }
  __syncthreads();

  // ---------------- scores = q @ k^T * scale, softmax over 24 joints --------
  {
    v8f sc[2]; sc[0] = vzero; sc[1] = vzero;
#pragma unroll
    for (int kk = 0; kk < 4; ++kk) {
      v16bf a0 = load_frag(qb[pos], LDA, 0,  kk * 32, lane);
      v16bf a1 = load_frag(qb[pos], LDA, 16, kk * 32, lane);
      v16bf bb = load_frag(kb[pos], LDA, nh * 16, kk * 32, lane);
      sc[0] = wmma_bf16(a0, bb, sc[0]);
      sc[1] = wmma_bf16(a1, bb, sc[1]);
    }
    const int jcol = nh * 16 + lrow;
#pragma unroll
    for (int mt = 0; mt < 2; ++mt)
#pragma unroll
      for (int r = 0; r < 8; ++r) {
        const int row = mt * 16 + r + 8 * lhi;
        if (row < KJ && jcol < KJ) sbuf[pos][row * LDSC + jcol] = sc[mt][r] * SCALEf;
      }
  }
  __syncthreads();

  if (lane < 12) {                          // 12 lanes/wave -> 24 rows/position
    const int row = lane * 2 + nh;
    const float* srow = &sbuf[pos][row * LDSC];
    float mx = srow[0];
    for (int j = 1; j < KJ; ++j) mx = fmaxf(mx, srow[j]);
    float sum = 0.f;
    for (int j = 0; j < KJ; ++j) sum += __expf(srow[j] - mx);
    const float inv = 1.f / sum;
    unsigned short* arow = &attnb[pos][row * LDATT];
    for (int j = 0; j < KJ; ++j) arow[j] = f2bf(__expf(srow[j] - mx) * inv);
    for (int j = KJ; j < 32; ++j) arow[j] = 0;   // zero K-padding (vs NaN leak)
  }
  __syncthreads();

  // ---------------- out1 = attn @ v  (single K=32 step) ---------------------
#pragma unroll
  for (int mt = 0; mt < 2; ++mt)
#pragma unroll
    for (int nt = 0; nt < 4; ++nt) acc[mt][nt] = vzero;
  {
    v16bf a0 = load_frag(attnb[pos], LDATT, 0,  0, lane);
    v16bf a1 = load_frag(attnb[pos], LDATT, 16, 0, lane);
#pragma unroll
    for (int nt = 0; nt < 4; ++nt) {
      v16bf bb = load_frag(vT[pos], LDV, (nh * 4 + nt) * 16, 0, lane);
      acc[0][nt] = wmma_bf16(a0, bb, acc[0][nt]);
      acc[1][nt] = wmma_bf16(a1, bb, acc[1][nt]);
    }
  }
#pragma unroll
  for (int mt = 0; mt < 2; ++mt)
#pragma unroll
    for (int nt = 0; nt < 4; ++nt) {
      const int gcol = (nh * 4 + nt) * 16 + lrow;
#pragma unroll
      for (int r = 0; r < 8; ++r) {
        const int row = mt * 16 + r + 8 * lhi;
        if (row < KJ) bufA[pos][row * LDA + gcol] = f2bf(acc[mt][nt][r]);
      }
    }
  __syncthreads();

  // ---------------- proj ----------------------------------------------------
  stage_w(W_proj, TDd, 0, wbuf, tid);
  __syncthreads();
#pragma unroll
  for (int mt = 0; mt < 2; ++mt)
#pragma unroll
    for (int nt = 0; nt < 4; ++nt) acc[mt][nt] = vzero;
  mm24(bufA[pos], LDA, wbuf, LDA, nh * 4, lane, acc);
#pragma unroll
  for (int mt = 0; mt < 2; ++mt)
#pragma unroll
    for (int nt = 0; nt < 4; ++nt) {
      const int gcol = (nh * 4 + nt) * 16 + lrow;
      const float bias = b_proj[gcol];
#pragma unroll
      for (int r = 0; r < 8; ++r) {
        const int row = mt * 16 + r + 8 * lhi;
        if (row < KJ) qb[pos][row * LDA + gcol] = f2bf(acc[mt][nt][r] + bias);
      }
    }
  __syncthreads();

  // ---------------- from + bias + residual -> global ------------------------
  stage_w(W_from, CPJd, 0, wbuf, tid);
  __syncthreads();
#pragma unroll
  for (int mt = 0; mt < 2; ++mt)
#pragma unroll
    for (int nt = 0; nt < 4; ++nt) acc[mt][nt] = vzero;
  mm24(qb[pos], LDA, wbuf, LDA, nh * 4, lane, acc);
#pragma unroll
  for (int mt = 0; mt < 2; ++mt)
#pragma unroll
    for (int nt = 0; nt < 4; ++nt) {
      const int gcol = (nh * 4 + nt) * 16 + lrow;
      const float bias = b_from[gcol];
#pragma unroll
      for (int r = 0; r < 8; ++r) {
        const int row = mt * 16 + r + 8 * lhi;
        if (row < KJ) {
          const size_t gi = ((size_t)(bB * Cdim + row * CPJd + gcol)) * Tdim + t;
          out[gi] = x[gi] + bias + acc[mt][nt][r];
        }
      }
    }
}

extern "C" void kernel_launch(void* const* d_in, const int* in_sizes, int n_in,
                              void* d_out, int out_size, void* d_ws, size_t ws_size,
                              hipStream_t stream) {
  (void)in_sizes; (void)n_in; (void)out_size; (void)d_ws; (void)ws_size;
  const float* x      = (const float*)d_in[0];
  const float* W_tok  = (const float*)d_in[1];
  const float* b_tok  = (const float*)d_in[2];
  const float* g_norm = (const float*)d_in[3];
  const float* W_qkv  = (const float*)d_in[4];
  const float* b_qkv  = (const float*)d_in[5];
  const float* W_proj = (const float*)d_in[6];
  const float* b_proj = (const float*)d_in[7];
  const float* W_from = (const float*)d_in[8];
  const float* b_from = (const float*)d_in[9];
  float* out = (float*)d_out;

  dim3 grid(Bdim * (Tdim / TT));
  dim3 block(NTHREADS);
  wan_joint_attn_kernel<<<grid, block, 0, stream>>>(
      x, W_tok, b_tok, g_norm, W_qkv, b_qkv, W_proj, b_proj, W_from, b_from, out);
}
